// RAND_model_34737695490360
// MI455X (gfx1250) — matrix operand order repaired and checked by reading, not compiled, not run on hardware
//
#include <hip/hip_runtime.h>
#include <hip/hip_bf16.h>

// ---------------- types for WMMA ----------------
typedef __attribute__((ext_vector_type(16))) __bf16 v16bf;
typedef __attribute__((ext_vector_type(8)))  float  v8f;

#define N_ROWS   100000
#define IN_DIM_  256
#define HID_DIM_ 512
#define OUT_DIM_ 128
#define SAMPLES  16
#define ANO_NUM  10000                 // int(N * 0.1)
#define POS_NUM  (N_ROWS - ANO_NUM)    // 90000
#define SORT_N   131072                // 2^17 >= N
#define NEG_SLOPE_ 0.01f
#define OUT_LDS  136                   // row stride (272 B = 17*16 B, 4-bank rot)

// ---------------- bf16 helpers ----------------
__device__ __forceinline__ unsigned short f2bf_us(float f) {
  unsigned u = __float_as_uint(f);
  unsigned r = (u + 0x7FFFu + ((u >> 16) & 1u)) >> 16;
  return (unsigned short)r;
}
__device__ __forceinline__ __bf16 f2bf(float f) {
  unsigned short h = f2bf_us(f);
  return __builtin_bit_cast(__bf16, h);
}
__device__ __forceinline__ float bf2f(__bf16 b) {
  unsigned short h = __builtin_bit_cast(unsigned short, b);
  return __uint_as_float(((unsigned)h) << 16);
}

// ---------------- WMMA fragment layout helpers ----------------
// A fragment (16x32, 16-bit): lanes 0-15 row M=lane, K in {0..7,16..23};
// lanes 16-31 row M=lane-16, K in {8..15,24..31}.
// Packed storage (global B / LDS hidden): tile(kt)*512 + lane*16 + i.
__device__ __forceinline__ int a_pack_idx(int m, int k) {
  int kt   = k >> 5;
  int kk   = k & 31;
  int lane = (((kk >> 3) & 1) << 4) + m;
  int i    = (kk & 7) | ((kk >> 4) << 3);
  return (kt << 9) + (lane << 4) + i;
}

// A fragment straight from f32 global (used once per row-tile, cold path)
__device__ __forceinline__ v16bf load_a_f32(const float* A, int lda, int row0,
                                            int k0, int lane) {
  const int m  = lane & 15;
  const int kb = (lane >> 4) << 3;  // 0 or 8
  const float* p = A + (size_t)(row0 + m) * lda + k0 + kb;
  v16bf a;
#pragma unroll
  for (int i = 0; i < 8; ++i) a[i] = f2bf(p[i]);
#pragma unroll
  for (int i = 0; i < 8; ++i) a[8 + i] = f2bf(p[16 + i]);
  return a;
}

// A fragment from plain row-major bf16 LDS (two ds_load_b128 per lane)
__device__ __forceinline__ v16bf load_a_row_lds(const __bf16* A, int ld,
                                                int row0, int k0, int lane) {
  const int m  = lane & 15;
  const int kb = (lane >> 4) << 3;  // 0 or 8
  const __bf16* p = A + (size_t)(row0 + m) * ld + k0 + kb;
  v16bf a;
#pragma unroll
  for (int i = 0; i < 8; ++i) a[i] = p[i];
#pragma unroll
  for (int i = 0; i < 8; ++i) a[8 + i] = p[16 + i];
  return a;
}

// ---------------- kernel 0: pack f32 weights into B-fragment layout --------
__global__ __launch_bounds__(256) void pack_b_kernel(
    const float* __restrict__ B, __bf16* __restrict__ Bp, int K, int Nn) {
  int e = blockIdx.x * blockDim.x + threadIdx.x;
  if (e >= K * Nn) return;
  int tile = e >> 9, idx = e & 511;
  int lane = idx >> 4, i = idx & 15;
  int ktiles = K >> 5;
  int nt = tile / ktiles, kt = tile - nt * ktiles;
  int n = (nt << 4) + (lane & 15);
  int k = (kt << 5) + ((i >> 3) << 4) + ((lane >> 4) << 3) + (i & 7);
  Bp[e] = f2bf(B[(size_t)k * Nn + n]);
}

// ---------------- kernel 1: center = leaky_relu(X@w1)@w2 ----------------
// Also emits a bf16 copy of center (consumed byte-for-byte by attention).
__global__ __launch_bounds__(96) void mlp_center_kernel(
    const float* __restrict__ feats, const __bf16* __restrict__ w1p,
    const __bf16* __restrict__ w2p, float* __restrict__ center,
    __bf16* __restrict__ centerbf) {
  __shared__ __align__(32) __bf16 hid[3][16 * HID_DIM_];  // 3 * 16 KB packed
  const int lane = threadIdx.x & 31;
  const int wave = threadIdx.x >> 5;
  const int tile = blockIdx.x * 3 + wave;
  if (tile * 16 >= N_ROWS) return;  // wave-uniform guard; no barriers used
  const int row0 = tile * 16;
  __bf16* h = hid[wave];

  v16bf afrag[IN_DIM_ / 32];
#pragma unroll
  for (int kt = 0; kt < IN_DIM_ / 32; ++kt)
    afrag[kt] = load_a_f32(feats, IN_DIM_, row0, kt * 32, lane);

  const int n  = lane & 15;
  const int mb = (lane >> 4) << 3;

  // stage 1: hid = leaky_relu(feats(16x256) @ w1(256x512)) -> packed LDS
  for (int nt = 0; nt < HID_DIM_ / 16; ++nt) {
    v8f acc = {};
    const __bf16* bp = w1p + ((size_t)nt * (IN_DIM_ / 32) << 9) + (lane << 4);
    __builtin_prefetch(bp + ((size_t)(IN_DIM_ / 32) << 9), 0, 1);
#pragma unroll
    for (int kt = 0; kt < IN_DIM_ / 32; ++kt) {
      v16bf b = *(const v16bf*)(bp + ((size_t)kt << 9));
      acc = __builtin_amdgcn_wmma_f32_16x16x32_bf16(false, afrag[kt], false, b,
                                                    (short)0, acc, false, false);
    }
#pragma unroll
    for (int r = 0; r < 8; ++r) {
      float v = acc[r];
      v = v > 0.0f ? v : NEG_SLOPE_ * v;
      h[a_pack_idx(mb + r, nt * 16 + n)] = f2bf(v);
    }
  }

  // stage 2: center = hid(16x512) @ w2(512x128)
  for (int nt = 0; nt < OUT_DIM_ / 16; ++nt) {
    v8f acc = {};
    const __bf16* bp = w2p + ((size_t)nt * (HID_DIM_ / 32) << 9) + (lane << 4);
    __builtin_prefetch(bp + ((size_t)(HID_DIM_ / 32) << 9), 0, 1);
    for (int kt = 0; kt < HID_DIM_ / 32; ++kt) {
      v16bf a = *(const v16bf*)(h + (kt << 9) + (lane << 4));
      v16bf b = *(const v16bf*)(bp + ((size_t)kt << 9));
      acc = __builtin_amdgcn_wmma_f32_16x16x32_bf16(false, a, false, b,
                                                    (short)0, acc, false, false);
    }
#pragma unroll
    for (int r = 0; r < 8; ++r) {
      size_t off = (size_t)(row0 + mb + r) * OUT_DIM_ + nt * 16 + n;
      center[off]   = acc[r];
      centerbf[off] = f2bf(acc[r]);
    }
  }
}

// ---------------- kernel 2: diff + sortable composite key ----------------
__global__ __launch_bounds__(256) void diff_key_kernel(
    const float* __restrict__ center, unsigned long long* __restrict__ keys) {
  int i = blockIdx.x * blockDim.x + threadIdx.x;
  if (i >= SORT_N) return;
  if (i >= N_ROWS) { keys[i] = ~0ull; return; }
  const float4* c4 = (const float4*)(center + (size_t)i * OUT_DIM_);
  float s = 0.0f;
  for (int j = 0; j < OUT_DIM_ / 4; ++j) {
    float4 v = c4[j];
    s += v.x; s += v.y; s += v.z; s += v.w;
  }
  float mean = s * (1.0f / OUT_DIM_);
  float d = 0.0f;
  for (int j = 0; j < OUT_DIM_ / 4; ++j) {
    float4 v = c4[j];
    d += (v.x - mean); d += (v.y - mean); d += (v.z - mean); d += (v.w - mean);
  }
  unsigned u = __float_as_uint(d);
  u = (u & 0x80000000u) ? ~u : (u | 0x80000000u);  // order-preserving map
  keys[i] = (((unsigned long long)u) << 32) | (unsigned)i;  // stable ties
}

// ---------------- kernel 3a: bitonic global step (j > 512) ----------------
__global__ __launch_bounds__(256) void bitonic_kernel(
    unsigned long long* __restrict__ keys, int j, int k) {
  int i = blockIdx.x * blockDim.x + threadIdx.x;
  int ixj = i ^ j;
  if (ixj > i) {
    unsigned long long a = keys[i];
    unsigned long long b = keys[ixj];
    bool asc = ((i & k) == 0);
    if ((a > b) == asc) { keys[i] = b; keys[ixj] = a; }
  }
}

// ---------------- kernel 3b: fused LDS bitonic tail (all j <= 512) --------
__global__ __launch_bounds__(512) void bitonic_fused_kernel(
    unsigned long long* __restrict__ keys, int k, int jstart) {
  __shared__ unsigned long long sk[1024];  // 8 KB
  const int t    = threadIdx.x;
  const int base = blockIdx.x * 1024;
  sk[t]       = keys[base + t];
  sk[t + 512] = keys[base + t + 512];
  __syncthreads();
  for (int j = jstart; j > 0; j >>= 1) {
    int i   = ((t & ~(j - 1)) << 1) | (t & (j - 1));
    int ixj = i | j;
    unsigned long long a = sk[i];
    unsigned long long b = sk[ixj];
    bool asc = (((base + i) & k) == 0);
    if ((a > b) == asc) { sk[i] = b; sk[ixj] = a; }
    __syncthreads();
  }
  keys[base + t]       = sk[t];
  keys[base + t + 512] = sk[t + 512];
}

// ---------------- kernel 4: attention over pos rows ----------------
// 4 pos rows / block -> 68 concat rows, padded to 80 (5 M-tiles).
// hcat staged via async global->LDS byte copies of bf16 center rows.
__global__ __launch_bounds__(256) void attention_kernel(
    const __bf16* __restrict__ centerbf, const __bf16* __restrict__ W1p,
    const __bf16* __restrict__ W2p, const int* __restrict__ nbr,
    const unsigned long long* __restrict__ keys, float* __restrict__ out) {
  __shared__ __align__(32) __bf16 hcat[80 * OUT_LDS];  // row-major, 21.8 KB
  __shared__ __align__(32) __bf16 scor[80 * OUT_LDS];  // row-major, 21.8 KB
  __shared__ __align__(32) __bf16 wsum[16 * OUT_LDS];  // row-major, 4.4 KB
  __shared__ int srcrow[80];
  __shared__ int prow[4];

  const int t    = threadIdx.x;
  const int lane = t & 31;
  const int wave = t >> 5;
  const int pbase = blockIdx.x * 4;  // grid = POS_NUM/4 exactly

  if (t < 4) prow[t] = (int)(unsigned)keys[pbase + t];  // low 32 bits = row id
  __syncthreads();
  if (t < 80) {
    int src = -1;
    if (t < 68) {
      int p = t / 17, s = t % 17;
      int g = prow[p];
      src = (s == 0) ? g : nbr[(size_t)g * SAMPLES + (s - 1)];
    }
    srcrow[t] = src;
  }
  __syncthreads();

  // async gather: 16 lanes x 16 B per row, direct global -> LDS (no VGPR trip)
  {
    const int sub = t & 15;  // 16-byte chunk within the 256 B row
    for (int r = (t >> 4); r < 68; r += 16) {
      int src = srcrow[r];
      unsigned long long ga =
          (unsigned long long)(const void*)(centerbf + (size_t)src * OUT_DIM_ +
                                            (sub << 3));
      unsigned lo = (unsigned)(unsigned long long)(const void*)(
          &hcat[r * OUT_LDS + (sub << 3)]);
      asm volatile("global_load_async_to_lds_b128 %0, %1, off"
                   :: "v"(lo), "v"(ga) : "memory");
    }
    // zero-fill padding rows 68..79 (plain stores)
    for (int e = t; e < 12 * (OUT_DIM_ / 4); e += 256) {
      int r = 68 + (e >> 5), c4 = (e & 31) << 2;
      uint2 z; z.x = 0u; z.y = 0u;
      *(uint2*)(&hcat[r * OUT_LDS + c4]) = z;
    }
    asm volatile("s_wait_asynccnt 0x0" ::: "memory");
  }
  __syncthreads();

  // scores = tanh(hcat(80x128) @ W1(128x128)) : 5 Mtiles x 8 Ntiles, 5/wave
  for (int tidx = wave; tidx < 40; tidx += 8) {
    int mt = tidx >> 3, nt = tidx & 7;
    v8f acc = {};
    const __bf16* bp = W1p + ((nt * 4) << 9) + (lane << 4);
#pragma unroll
    for (int kt = 0; kt < 4; ++kt) {
      v16bf a = load_a_row_lds(hcat, OUT_LDS, mt * 16, kt * 32, lane);
      v16bf b = *(const v16bf*)(bp + (kt << 9));
      acc = __builtin_amdgcn_wmma_f32_16x16x32_bf16(false, a, false, b,
                                                    (short)0, acc, false, false);
    }
    const int n = lane & 15, mb = (lane >> 4) << 3;
#pragma unroll
    for (int r = 0; r < 8; ++r)
      scor[(mt * 16 + mb + r) * OUT_LDS + nt * 16 + n] = f2bf(tanhf(acc[r]));
  }
  __syncthreads();

  // wsum[p][c] = sum_s scores*hcat over 17 concat rows; rows 4..15 zero pad
  for (int e = t; e < 16 * OUT_DIM_; e += 256) {
    int p = e >> 7, c = e & 127;
    float acc = 0.0f;
    if (p < 4) {
#pragma unroll
      for (int s = 0; s < 17; ++s) {
        int r = (p * 17 + s) * OUT_LDS + c;
        acc += bf2f(scor[r]) * bf2f(hcat[r]);
      }
    }
    wsum[p * OUT_LDS + c] = f2bf(acc);
  }
  __syncthreads();

  // agg = wsum(16x128) @ W2(128x128): 8 N-tiles, one per wave; scatter rows 0..3
  {
    int nt = wave;
    v8f acc = {};
    const __bf16* bp = W2p + ((nt * 4) << 9) + (lane << 4);
#pragma unroll
    for (int kt = 0; kt < 4; ++kt) {
      v16bf a = load_a_row_lds(wsum, OUT_LDS, 0, kt * 32, lane);
      v16bf b = *(const v16bf*)(bp + (kt << 9));
      acc = __builtin_amdgcn_wmma_f32_16x16x32_bf16(false, a, false, b,
                                                    (short)0, acc, false, false);
    }
    const int n = lane & 15, mb = (lane >> 4) << 3;
#pragma unroll
    for (int r = 0; r < 8; ++r) {
      int p = mb + r;
      if (p < 4)
        out[(size_t)prow[p] * OUT_DIM_ + nt * 16 + n] = acc[r];
    }
  }
}

// ---------------- kernel 5: neg rows copy center -> out (float4) ----------
__global__ __launch_bounds__(256) void neg_copy_kernel(
    const float* __restrict__ center,
    const unsigned long long* __restrict__ keys, float* __restrict__ out) {
  int e = blockIdx.x * blockDim.x + threadIdx.x;
  if (e >= ANO_NUM * (OUT_DIM_ / 4)) return;
  int p = POS_NUM + (e >> 5);               // sorted position in neg region
  int c4 = (e & 31) << 2;
  int row = (int)(unsigned)keys[p];
  *(float4*)(out + (size_t)row * OUT_DIM_ + c4) =
      *(const float4*)(center + (size_t)row * OUT_DIM_ + c4);
}

// ---------------- host launcher ----------------
extern "C" void kernel_launch(void* const* d_in, const int* in_sizes, int n_in,
                              void* d_out, int out_size, void* d_ws,
                              size_t ws_size, hipStream_t stream) {
  (void)in_sizes; (void)n_in; (void)out_size; (void)ws_size;
  const float* feats = (const float*)d_in[0];
  const float* w1    = (const float*)d_in[1];
  const float* w2    = (const float*)d_in[2];
  const float* W1    = (const float*)d_in[3];
  const float* W2    = (const float*)d_in[4];
  const int*   nbr   = (const int*)d_in[5];
  float* out = (float*)d_out;

  // workspace carve-up (all offsets 32B aligned)
  char* ws = (char*)d_ws;
  float* center = (float*)ws;
  ws += (size_t)N_ROWS * OUT_DIM_ * sizeof(float);              // 51.2 MB
  __bf16* centerbf = (__bf16*)ws;
  ws += (size_t)N_ROWS * OUT_DIM_ * 2;                          // 25.6 MB
  unsigned long long* keys = (unsigned long long*)ws;
  ws += (size_t)SORT_N * sizeof(unsigned long long);            // 1 MB
  __bf16* w1p = (__bf16*)ws; ws += (size_t)IN_DIM_ * HID_DIM_ * 2;
  __bf16* w2p = (__bf16*)ws; ws += (size_t)HID_DIM_ * OUT_DIM_ * 2;
  __bf16* W1p = (__bf16*)ws; ws += (size_t)OUT_DIM_ * OUT_DIM_ * 2;
  __bf16* W2p = (__bf16*)ws; ws += (size_t)OUT_DIM_ * OUT_DIM_ * 2;

  // 0) pre-swizzle weights into WMMA B-fragment layout (bf16)
  pack_b_kernel<<<(IN_DIM_ * HID_DIM_ + 255) / 256, 256, 0, stream>>>(w1, w1p, IN_DIM_, HID_DIM_);
  pack_b_kernel<<<(HID_DIM_ * OUT_DIM_ + 255) / 256, 256, 0, stream>>>(w2, w2p, HID_DIM_, OUT_DIM_);
  pack_b_kernel<<<(OUT_DIM_ * OUT_DIM_ + 255) / 256, 256, 0, stream>>>(W1, W1p, OUT_DIM_, OUT_DIM_);
  pack_b_kernel<<<(OUT_DIM_ * OUT_DIM_ + 255) / 256, 256, 0, stream>>>(W2, W2p, OUT_DIM_, OUT_DIM_);

  // 1) center (+ bf16 copy)
  const int tiles = N_ROWS / 16;                      // 6250 exact
  mlp_center_kernel<<<(tiles + 2) / 3, 96, 0, stream>>>(feats, w1p, w2p, center, centerbf);

  // 2) keys
  diff_key_kernel<<<SORT_N / 256, 256, 0, stream>>>(center, keys);

  // 3) bitonic sort: global passes for j > 512, fused LDS tail (45 launches)
  for (int k = 2; k <= SORT_N; k <<= 1) {
    int j = k >> 1;
    for (; j > 512; j >>= 1)
      bitonic_kernel<<<SORT_N / 256, 256, 0, stream>>>(keys, j, k);
    bitonic_fused_kernel<<<SORT_N / 1024, 512, 0, stream>>>(keys, k, j);
  }

  // 4) attention over pos rows (90000 / 4 = 22500 blocks exact)
  attention_kernel<<<POS_NUM / 4, 256, 0, stream>>>(centerbf, W1p, W2p, nbr, keys, out);

  // 5) neg rows passthrough
  neg_copy_kernel<<<(ANO_NUM * (OUT_DIM_ / 4) + 255) / 256, 256, 0, stream>>>(center, keys, out);
}
